// BitwiseTasNetBlock_27341761806687
// MI455X (gfx1250) — matrix operand order repaired and checked by reading, not compiled, not run on hardware
//
#include <hip/hip_runtime.h>
#include <hip/hip_bf16.h>

// ---------------------------------------------------------------------------
// BitwiseTasNet block for MI455X (gfx1250, wave32, WMMA).
//   activations: time-major [N][C] fp16, N = B*T = 65536, C in {256, 512}
//   weights:     binarized to fp16 (+/-1), [M][K] row-major
// GEMMs use v_wmma_f32_16x16x32_f16 with ISA-documented fragment layouts.
// BN2 folded into GEMM2 weights; BN1 stats fused into GEMM1 epilogue
// (per-j butterfly reduction). B-tile staging uses async-load-to-LDS when
// the toolchain exposes the builtin.
// ---------------------------------------------------------------------------

typedef _Float16 v8h  __attribute__((ext_vector_type(8)));
typedef _Float16 v16h __attribute__((ext_vector_type(16)));
typedef float    v8f  __attribute__((ext_vector_type(8)));
typedef int      v4i  __attribute__((vector_size(4 * sizeof(int))));

#define NTOT 65536   // B*T
#define TLEN 4096
#define CB_C 256
#define D_C  512
#define L_C  4

#if defined(__has_builtin)
#if __has_builtin(__builtin_amdgcn_global_load_async_to_lds_b128) && \
    __has_builtin(__builtin_amdgcn_s_wait_asynccnt)
#define HAVE_ASYNC_LDS 1
#endif
#endif

static __device__ __forceinline__ v16h cat8(v8h lo, v8h hi) {
  return __builtin_shufflevector(lo, hi, 0,1,2,3,4,5,6,7,8,9,10,11,12,13,14,15);
}

// --------------------------- prep kernels ----------------------------------

__global__ __launch_bounds__(256) void binarize_kernel(
    const float* __restrict__ w, _Float16* __restrict__ ws, int n) {
  int i = blockIdx.x * 256 + threadIdx.x;
  if (i < n) ws[i] = (_Float16)copysignf(1.0f, w[i]);
}

// x [B][CB][T] fp32 -> xc [N][CB] fp16 (time-major), LDS tile transpose.
// Block: 64(c) x 64(t) tile; reads coalesced along t, writes coalesced along c.
__global__ __launch_bounds__(256) void transpose_in_kernel(
    const float* __restrict__ x, _Float16* __restrict__ xc) {
  __shared__ float sh[64][65];
  const int bt = blockIdx.x & 63;          // T/64 = 64 tiles
  const int bc = (blockIdx.x >> 6) & 3;    // CB/64 = 4 tiles
  const int bb = blockIdx.x >> 8;          // batch
  const int t0 = bt * 64, c0 = bc * 64;
  const int tid = threadIdx.x;
#pragma unroll
  for (int i = 0; i < 16; ++i) {
    const int ii = tid + i * 256;
    const int cs = ii >> 6, ts = ii & 63;
    sh[cs][ts] = x[((size_t)(bb * CB_C + c0 + cs) << 12) + t0 + ts];
  }
  __syncthreads();
#pragma unroll
  for (int i = 0; i < 16; ++i) {
    const int ii = tid + i * 256;
    const int ts = ii >> 6, cs = ii & 63;
    xc[((size_t)(bb << 12) + t0 + ts) * CB_C + c0 + cs] = (_Float16)sh[cs][ts];
  }
}

// --------------------------- WMMA GEMM -------------------------------------
// D[m,n] = sum_k A[m,k]*Bact[n,k] (+ bias), M = MT*128, K = KH*32, N block=64.
// STATS: fused PReLU + BN-stat accumulation (GEMM1).
// FINAL: write fp32 out in [B][CB][T] layout with residual (last GEMM2).

template<int KH, int MT, bool STATS, bool FINAL>
__global__ __launch_bounds__(256) void gemm_bin(
    const _Float16* __restrict__ A, const _Float16* __restrict__ Bact,
    const float* __restrict__ bias, const float* __restrict__ aprelu,
    _Float16* __restrict__ Z, float* __restrict__ gsum, float* __restrict__ gsq,
    float* __restrict__ Out, const float* __restrict__ Xres) {
  constexpr int K = KH * 32;
  constexpr int M = MT * 128;
  extern __shared__ _Float16 sB[];                 // [64][K] activation tile
  __shared__ float sStat[STATS ? 2 * M : 1];

  const int tid = threadIdx.x;
  const int nb0 = blockIdx.x * 64;

  // stage 64 activation rows (contiguous in [N][K]) into LDS
#ifdef HAVE_ASYNC_LDS
  {
    typedef __attribute__((address_space(1))) v4i gvec;
    typedef __attribute__((address_space(3))) v4i lvec;
    gvec* gsrc = (gvec*)(Bact + (size_t)nb0 * K);
    lvec* ldst = (lvec*)sB;
    constexpr int cnt = 64 * K / 8;                // 16B chunks
#pragma unroll
    for (int i = 0; i < cnt / 256; ++i) {
      const int e = i * 256 + tid;
      __builtin_amdgcn_global_load_async_to_lds_b128(gsrc + e, ldst + e, 0, 0);
    }
    __builtin_amdgcn_s_wait_asynccnt(0);
  }
#else
  {
    const uint4* src = reinterpret_cast<const uint4*>(Bact + (size_t)nb0 * K);
    uint4* dst = reinterpret_cast<uint4*>(sB);
    constexpr int cnt = 64 * K / 8;
#pragma unroll
    for (int i = 0; i < cnt / 256; ++i) dst[i * 256 + tid] = src[i * 256 + tid];
  }
#endif
  if constexpr (STATS) {
    for (int i = tid; i < 2 * M; i += 256) sStat[i] = 0.f;
  }
  __syncthreads();

  const int wave = tid >> 5, lane = tid & 31;
  const int lrow = lane & 15, kh = lane >> 4;
  const float aval = STATS ? aprelu[0] : 0.f;

  for (int j = 0; j < MT; ++j) {
    const int mb = (wave * MT + j) * 16;
    // A fragments: lane lrow holds row mb+lrow; elems 0..7 = K kh*8..+8,
    // elems 8..15 = K 16+kh*8..+8 (ISA 16-bit A 16x32 layout)
    v16h af[KH];
    const _Float16* ap = A + (size_t)(mb + lrow) * K + kh * 8;
#pragma unroll
    for (int kf = 0; kf < KH; ++kf) {
      v8h lo = *reinterpret_cast<const v8h*>(ap + kf * 32);
      v8h hi = *reinterpret_cast<const v8h*>(ap + kf * 32 + 16);
      af[kf] = cat8(lo, hi);
    }
    const int mbl = mb + kh * 8;                   // 8 consecutive M per lane
    float ssum[8] = {0,0,0,0,0,0,0,0}, ssq[8] = {0,0,0,0,0,0,0,0};
    for (int nt = 0; nt < 4; ++nt) {
      v8f c = {};
      // B fragment: lane holds column nt*16+lrow, elems 0..7 = K kh*8..+8, 8..15 = K 16+kh*8..+8
      const _Float16* bp = sB + (size_t)(nt * 16 + lrow) * K + kh * 8;
#pragma unroll
      for (int kf = 0; kf < KH; ++kf) {
        v8h blo = *reinterpret_cast<const v8h*>(bp + kf * 32);
        v8h bhi = *reinterpret_cast<const v8h*>(bp + kf * 32 + 16);
        c = __builtin_amdgcn_wmma_f32_16x16x32_f16(
            false, af[kf], false, cat8(blo, bhi), (short)0, c, false, false);
      }
      const int n = nb0 + nt * 16 + lrow;
      if constexpr (!FINAL) {
        v8h zh;
#pragma unroll
        for (int r = 0; r < 8; ++r) {
          float v = c[r] + bias[mbl + r];
          if constexpr (STATS) {
            v = (v >= 0.f) ? v : aval * v;
            ssum[r] += v;
            ssq[r] += v * v;
          }
          zh[r] = (_Float16)v;
        }
        *reinterpret_cast<v8h*>(Z + (size_t)n * M + mbl) = zh;
      } else {
        const int bb = n >> 12, tt = n & (TLEN - 1);
#pragma unroll
        for (int r = 0; r < 8; ++r) {
          const int m = mbl + r;
          const size_t o = (((size_t)bb * CB_C + m) << 12) + tt;
          Out[o] = c[r] + bias[m] + Xres[o];
        }
      }
    }
    if constexpr (STATS) {
      // lanes 0..15 (and 16..31) share the same 8 channels: one butterfly per j
#pragma unroll
      for (int r = 0; r < 8; ++r) {
        float s = ssum[r], q = ssq[r];
#pragma unroll
        for (int mm = 1; mm < 16; mm <<= 1) {
          s += __shfl_xor(s, mm, 32);
          q += __shfl_xor(q, mm, 32);
        }
        if (lrow == 0) {
          atomicAdd(&sStat[mbl + r], s);
          atomicAdd(&sStat[M + mbl + r], q);
        }
      }
    }
  }
  if constexpr (STATS) {
    __syncthreads();
    for (int i = tid; i < M; i += 256) {
      atomicAdd(&gsum[i], sStat[i]);
      atomicAdd(&gsq[i],  sStat[M + i]);
    }
  }
}

// ------------------ depthwise dilated causal conv + BN1 + PReLU ------------
// Reads z1 [N][512] fp16, applies zn = s1*z+t1 (zero pad), y = sum sign(wd[j])
// * zn[t-(2-j)*dil] + bd, z2 = prelu(y); accumulates BN2 stats.

__global__ __launch_bounds__(256) void dwconv_kernel(
    const _Float16* __restrict__ Z1, const float* __restrict__ s1,
    const float* __restrict__ t1, const float* __restrict__ wd,
    const float* __restrict__ bd, const float* __restrict__ a2p, int dil,
    _Float16* __restrict__ Z2, float* __restrict__ gsum, float* __restrict__ gsq) {
  __shared__ float sStat[2 * D_C];
  const int tid = threadIdx.x;
  for (int i = tid; i < 2 * D_C; i += 256) sStat[i] = 0.f;
  __syncthreads();

  const int g = tid & 63, sr = tid >> 6;
  const int d0 = g * 8;
  float sc[8], sh[8], w0[8], w1[8], w2[8], bv[8];
#pragma unroll
  for (int i = 0; i < 8; ++i) {
    sc[i] = s1[d0 + i]; sh[i] = t1[d0 + i];
    w0[i] = copysignf(1.f, wd[(d0 + i) * 3 + 0]);
    w1[i] = copysignf(1.f, wd[(d0 + i) * 3 + 1]);
    w2[i] = copysignf(1.f, wd[(d0 + i) * 3 + 2]);
    bv[i] = bd[d0 + i];
  }
  const float a2 = a2p[0];
  float asum[8] = {0,0,0,0,0,0,0,0}, asq[8] = {0,0,0,0,0,0,0,0};
  const int n0 = blockIdx.x * 64;
  for (int jj = 0; jj < 16; ++jj) {
    const int n = n0 + sr + jj * 4;
    const int t = n & (TLEN - 1);
    const bool v1 = (t >= dil), v2 = (t >= 2 * dil);
    v8h x0 = *reinterpret_cast<const v8h*>(Z1 + (size_t)n * D_C + d0);
    v8h x1 = {}, x2 = {};
    if (v1) x1 = *reinterpret_cast<const v8h*>(Z1 + (size_t)(n - dil) * D_C + d0);
    if (v2) x2 = *reinterpret_cast<const v8h*>(Z1 + (size_t)(n - 2 * dil) * D_C + d0);
    v8h zo;
#pragma unroll
    for (int i = 0; i < 8; ++i) {
      float zn0 = sc[i] * (float)x0[i] + sh[i];
      float zn1 = v1 ? (sc[i] * (float)x1[i] + sh[i]) : 0.f;
      float zn2 = v2 ? (sc[i] * (float)x2[i] + sh[i]) : 0.f;
      float y = w0[i] * zn2 + w1[i] * zn1 + w2[i] * zn0 + bv[i];
      float z = (y >= 0.f) ? y : a2 * y;
      zo[i] = (_Float16)z;
      asum[i] += z; asq[i] += z * z;
    }
    *reinterpret_cast<v8h*>(Z2 + (size_t)n * D_C + d0) = zo;
  }
#pragma unroll
  for (int i = 0; i < 8; ++i) {
    atomicAdd(&sStat[d0 + i], asum[i]);
    atomicAdd(&sStat[D_C + d0 + i], asq[i]);
  }
  __syncthreads();
  for (int i = tid; i < D_C; i += 256) {
    atomicAdd(&gsum[i], sStat[i]);
    atomicAdd(&gsq[i],  sStat[D_C + i]);
  }
}

// ----------------------- BN stat finalize: scale/shift ---------------------

__global__ __launch_bounds__(256) void bn_finalize_kernel(
    const float* __restrict__ gsum, const float* __restrict__ gsq,
    const float* __restrict__ g, const float* __restrict__ be,
    float* __restrict__ s, float* __restrict__ t) {
  const int d = blockIdx.x * 256 + threadIdx.x;
  if (d >= D_C) return;
  const float invN = 1.0f / (float)NTOT;
  const float mean = gsum[d] * invN;
  const float var = gsq[d] * invN - mean * mean;
  const float sv = g[d] * rsqrtf(var + 1e-5f);
  s[d] = sv;
  t[d] = be[d] - mean * sv;
}

// ------------- fold BN2 into GEMM2 weights: W' = sign(w2)*s2, bias' --------

__global__ __launch_bounds__(256) void fold_w2_kernel(
    const float* __restrict__ w2, const float* __restrict__ s2,
    const float* __restrict__ t2, const float* __restrict__ b2,
    _Float16* __restrict__ weff, float* __restrict__ beff) {
  __shared__ float red[256];
  const int cb = blockIdx.x, tid = threadIdx.x;
  float acc = 0.f;
  for (int d = tid; d < D_C; d += 256) {
    const float sgn = copysignf(1.f, w2[(size_t)cb * D_C + d]);
    weff[(size_t)cb * D_C + d] = (_Float16)(sgn * s2[d]);
    acc += sgn * t2[d];
  }
  red[tid] = acc; __syncthreads();
  for (int o = 128; o > 0; o >>= 1) {
    if (tid < o) red[tid] += red[tid + o];
    __syncthreads();
  }
  if (tid == 0) beff[cb] = b2[cb] + red[0];
}

// --------------------------------- host ------------------------------------

extern "C" void kernel_launch(void* const* d_in, const int* in_sizes, int n_in,
                              void* d_out, int out_size, void* d_ws, size_t ws_size,
                              hipStream_t stream) {
  (void)in_sizes; (void)n_in; (void)out_size; (void)ws_size;
  const float* x   = (const float*)d_in[0];
  const float* w1  = (const float*)d_in[1];
  const float* b1  = (const float*)d_in[2];
  const float* a1  = (const float*)d_in[3];
  const float* g1  = (const float*)d_in[4];
  const float* be1 = (const float*)d_in[5];
  const float* wd  = (const float*)d_in[6];
  const float* bd  = (const float*)d_in[7];
  const float* a2  = (const float*)d_in[8];
  const float* g2  = (const float*)d_in[9];
  const float* be2 = (const float*)d_in[10];
  const float* w2  = (const float*)d_in[11];
  const float* b2  = (const float*)d_in[12];

  char* wsb = (char*)d_ws;
  size_t off = 0;
  auto take = [&](size_t bytes) -> char* {
    char* p = wsb + off;
    off = (off + bytes + 255) & ~(size_t)255;
    return p;
  };
  _Float16* xcur = (_Float16*)take((size_t)NTOT * CB_C * 2);       // 32 MiB
  _Float16* z1   = (_Float16*)take((size_t)NTOT * D_C * 2);        // 64 MiB
  _Float16* z2   = (_Float16*)take((size_t)NTOT * D_C * 2);        // 64 MiB
  _Float16* w1s  = (_Float16*)take((size_t)L_C * D_C * CB_C * 2);  // 1 MiB
  _Float16* weff = (_Float16*)take((size_t)CB_C * D_C * 2);        // 256 KiB
  float* beff    = (float*)take(CB_C * 4);
  float* stats   = (float*)take(4 * D_C * 4);
  float* gsum1 = stats, *gsq1 = stats + D_C;
  float* gsum2 = stats + 2 * D_C, *gsq2 = stats + 3 * D_C;
  float* sbuf  = (float*)take(4 * D_C * 4);
  float* s1 = sbuf, *t1 = sbuf + D_C, *s2 = sbuf + 2 * D_C, *t2 = sbuf + 3 * D_C;

  binarize_kernel<<<(L_C * D_C * CB_C) / 256, 256, 0, stream>>>(
      w1, w1s, L_C * D_C * CB_C);
  transpose_in_kernel<<<16 * 4 * 64, 256, 0, stream>>>(x, xcur);

  for (int l = 0; l < L_C; ++l) {
    (void)hipMemsetAsync(stats, 0, 4 * D_C * 4, stream);
    // 1x1 conv CB->D (WMMA) + bias + PReLU + BN1 stats
    gemm_bin<8, 4, true, false><<<NTOT / 64, 256, 64 * CB_C * 2, stream>>>(
        w1s + (size_t)l * D_C * CB_C, xcur, b1 + l * D_C, a1 + l,
        z1, gsum1, gsq1, nullptr, nullptr);
    bn_finalize_kernel<<<2, 256, 0, stream>>>(gsum1, gsq1, g1 + l * D_C,
                                              be1 + l * D_C, s1, t1);
    // depthwise dilated causal conv (BN1 applied inline) + PReLU + BN2 stats
    dwconv_kernel<<<NTOT / 64, 256, 0, stream>>>(
        z1, s1, t1, wd + (size_t)l * D_C * 3, bd + l * D_C, a2 + l,
        1 << l, z2, gsum2, gsq2);
    bn_finalize_kernel<<<2, 256, 0, stream>>>(gsum2, gsq2, g2 + l * D_C,
                                              be2 + l * D_C, s2, t2);
    fold_w2_kernel<<<CB_C, 256, 0, stream>>>(w2 + (size_t)l * CB_C * D_C, s2,
                                             t2, b2 + l * CB_C, weff, beff);
    // 1x1 conv D->CB (WMMA) with BN2 folded into weights
    if (l < L_C - 1) {
      gemm_bin<16, 2, false, false><<<NTOT / 64, 256, 64 * D_C * 2, stream>>>(
          weff, z2, beff, nullptr, xcur, nullptr, nullptr, nullptr, nullptr);
    } else {
      gemm_bin<16, 2, false, true><<<NTOT / 64, 256, 64 * D_C * 2, stream>>>(
          weff, z2, beff, nullptr, nullptr, nullptr, nullptr,
          (float*)d_out, x);
    }
  }
}
